// GAT_14551349199566
// MI455X (gfx1250) — matrix operand order repaired and checked by reading, not compiled, not run on hardware
//
#include <hip/hip_runtime.h>

// ---------------------------------------------------------------------------
// Types for CDNA5 WMMA (wave32)
// ---------------------------------------------------------------------------
typedef __bf16 bf16_t;
typedef __attribute__((ext_vector_type(16))) __bf16 v16bf;
typedef __attribute__((ext_vector_type(8)))  __bf16 v8bf;
typedef __attribute__((ext_vector_type(8)))  float  v8f;

// ---------------------------------------------------------------------------
// Problem constants (from reference)
// ---------------------------------------------------------------------------
#define GN    10000            // nodes
#define GEDGE 160000           // raw edges
#define GET   (GEDGE + GN)     // edges incl. self loops
#define NEG_SLOPE 0.2f

// ---------------------------------------------------------------------------
// GEMM: C[M,Nc] = A[M,K] @ Bt^T, where Bt is B pre-transposed to [Nc,K].
// bf16 inputs, f32 accumulation via v_wmma_f32_16x16x32_bf16.
// Block: 256 threads = 8 waves, block tile 128x128, K staged 32 at a time in
// double-buffered LDS. Waves form a 4x2 grid; each wave computes a 32x64
// sub-tile as a 2x4 outer product of WMMA fragments held in distinct
// registers, so all 12 ds_loads issue before a single dscnt wait and the 8
// WMMAs go back-to-back into the matrix pipes.
// ---------------------------------------------------------------------------
#define BM 128
#define BN 128
#define BK 32

__global__ __launch_bounds__(256)
void gat_gemm_bf16(const bf16_t* __restrict__ A, const bf16_t* __restrict__ Bt,
                   float* __restrict__ C, int M, int K, int Nc) {
  __shared__ bf16_t As[2][BM][BK];   // 2 x 8 KB
  __shared__ bf16_t Bs[2][BN][BK];   // 2 x 8 KB, Bs[buf][col][k]

  const int tid  = threadIdx.x;
  const int lane = tid & 31;
  const int wave = tid >> 5;
  const int m0 = blockIdx.x * BM;
  const int n0 = blockIdx.y * BN;
  const int lane15 = lane & 15;
  const int half   = lane >> 4;
  const int wr = (wave & 3) * 32;    // wave's row offset within block tile
  const int wc = (wave >> 2) * 64;   // wave's col offset within block tile

  v8f acc[2][4];
#pragma unroll
  for (int i = 0; i < 2; ++i)
#pragma unroll
    for (int j = 0; j < 4; ++j)
#pragma unroll
      for (int q = 0; q < 8; ++q) acc[i][j][q] = 0.0f;

  const int KT = K / BK;

  // Stage tile kt into LDS buffer buf: all 16-byte contiguous transfers.
  auto stage = [&](int kt, int buf) {
    const int k0 = kt * BK;
    if (k0 + BK < K) {  // light prefetch of the following K-tile
      __builtin_prefetch(A + (size_t)(m0 + (tid >> 1)) * K + k0 + BK, 0, 1);
      __builtin_prefetch(Bt + (size_t)(n0 + (tid >> 1)) * K + k0 + BK, 0, 1);
    }
    // A tile (BM x BK): 512 chunks of 8 bf16, 2 per thread
#pragma unroll
    for (int it = 0; it < 2; ++it) {
      int c   = tid + it * 256;       // 0..511
      int row = c >> 2;               // 0..127
      int kk  = (c & 3) * 8;          // 0,8,16,24
      int gr  = m0 + row;
      v8bf va;
      if (gr < M) {
        va = *(const v8bf*)(A + (size_t)gr * K + k0 + kk);
      } else {
#pragma unroll
        for (int i = 0; i < 8; ++i) va[i] = (bf16_t)0.0f;
      }
      *(v8bf*)(&As[buf][row][kk]) = va;
    }
    // B tile (BN cols x BK): Bt is [Nc,K] so these are contiguous loads+stores
#pragma unroll
    for (int it = 0; it < 2; ++it) {
      int c   = tid + it * 256;       // 0..511
      int col = c >> 2;               // 0..127
      int kk  = (c & 3) * 8;
      v8bf vb = *(const v8bf*)(Bt + (size_t)(n0 + col) * K + k0 + kk);
      *(v8bf*)(&Bs[buf][col][kk]) = vb;
    }
  };

  stage(0, 0);
  __syncthreads();

  for (int kt = 0; kt < KT; ++kt) {
    const int buf = kt & 1;
    if (kt + 1 < KT) stage(kt + 1, buf ^ 1);

    // Preload ALL fragments into distinct registers (A: lane L holds row
    // (L&15), K halves split by L>>4; B: lane L holds col (L&15), K chunks
    // (L>>4)*8 and 16+(L>>4)*8, same order as A).
    v16bf af[2], bfr[4];
#pragma unroll
    for (int i = 0; i < 2; ++i) {
      const int arow = wr + i * 16 + lane15;
      v8bf lo = *(const v8bf*)(&As[buf][arow][half * 8]);
      v8bf hi = *(const v8bf*)(&As[buf][arow][16 + half * 8]);
#pragma unroll
      for (int q = 0; q < 8; ++q) { af[i][q] = lo[q]; af[i][8 + q] = hi[q]; }
    }
#pragma unroll
    for (int j = 0; j < 4; ++j) {
      const int bcol = wc + j * 16 + lane15;
      v8bf lo = *(const v8bf*)(&Bs[buf][bcol][half * 8]);
      v8bf hi = *(const v8bf*)(&Bs[buf][bcol][16 + half * 8]);
#pragma unroll
      for (int q = 0; q < 8; ++q) { bfr[j][q] = lo[q]; bfr[j][8 + q] = hi[q]; }
    }

    // 2x4 outer product: 8 independent back-to-back WMMAs
#pragma unroll
    for (int i = 0; i < 2; ++i)
#pragma unroll
      for (int j = 0; j < 4; ++j)
        acc[i][j] = __builtin_amdgcn_wmma_f32_16x16x32_bf16(
            /*neg_a=*/false, af[i], /*neg_b=*/false, bfr[j],
            /*c_mod=*/(short)0, acc[i][j], /*reuse_a=*/false, /*reuse_b=*/false);

    __syncthreads();
  }

  // Store D (C/D layout: VGPR q -> row base + 8*(L>>4) + q, col = L&15)
#pragma unroll
  for (int i = 0; i < 2; ++i) {
    const int r0 = m0 + wr + i * 16 + half * 8;
#pragma unroll
    for (int j = 0; j < 4; ++j) {
      const int cb = n0 + wc + j * 16 + lane15;
#pragma unroll
      for (int q = 0; q < 8; ++q) {
        int r = r0 + q;
        if (r < M) C[(size_t)r * Nc + cb] = acc[i][j][q];
      }
    }
  }
}

// ---------------------------------------------------------------------------
// Conversion kernels
// ---------------------------------------------------------------------------
__global__ void gat_cvt_bf16(const float* __restrict__ x, bf16_t* __restrict__ y,
                             long long n4) {   // n4 = n/4, n divisible by 4
  long long i = (long long)blockIdx.x * blockDim.x + threadIdx.x;
  if (i >= n4) return;
  float4 v = ((const float4*)x)[i];
  bf16_t* yp = y + i * 4;
  yp[0] = (bf16_t)v.x; yp[1] = (bf16_t)v.y; yp[2] = (bf16_t)v.z; yp[3] = (bf16_t)v.w;
}

// Wt[n*K + k] = bf16(W[k*Nc + n])  (transpose so GEMM B staging is contiguous)
__global__ void gat_cvt_w_trans(const float* __restrict__ W, bf16_t* __restrict__ Wt,
                                int K, int Nc) {
  long long t = (long long)blockIdx.x * blockDim.x + threadIdx.x;
  long long tot = (long long)K * Nc;
  if (t >= tot) return;
  int n = (int)(t % Nc);
  int k = (int)(t / Nc);
  Wt[(size_t)n * K + k] = (bf16_t)W[t];
}

__global__ void gat_fill(float* __restrict__ p, float v, long long n) {
  long long i = (long long)blockIdx.x * blockDim.x + threadIdx.x;
  if (i < n) p[i] = v;
}

// ---------------------------------------------------------------------------
// Attention pipeline (Hh, Cc compile-time powers of two -> shift/mask math)
// ---------------------------------------------------------------------------
template <int Hh, int Cc>
__global__ void gat_attn_logits(const float* __restrict__ XL,
                                const float* __restrict__ a_s,
                                const float* __restrict__ a_d,
                                float* __restrict__ als, float* __restrict__ ald) {
  int t = blockIdx.x * blockDim.x + threadIdx.x;
  if (t >= GN * Hh) return;
  int n = t / Hh, h = t & (Hh - 1);
  const float4* xp = (const float4*)(XL + ((size_t)n * Hh + h) * Cc);
  const float4* sp = (const float4*)(a_s + (size_t)h * Cc);
  const float4* dp = (const float4*)(a_d + (size_t)h * Cc);
  float s1 = 0.f, s2 = 0.f;
#pragma unroll
  for (int c = 0; c < Cc / 4; ++c) {
    float4 xv = xp[c], sv = sp[c], dv = dp[c];
    s1 += xv.x * sv.x + xv.y * sv.y + xv.z * sv.z + xv.w * sv.w;
    s2 += xv.x * dv.x + xv.y * dv.y + xv.z * dv.z + xv.w * dv.w;
  }
  als[t] = s1; ald[t] = s2;
}

__device__ __forceinline__ void gat_edge_nodes(const long long* __restrict__ ei,
                                               int e, int& s, int& d) {
  if (e < GEDGE) { s = (int)ei[e]; d = (int)ei[GEDGE + e]; }
  else           { s = e - GEDGE; d = e - GEDGE; }      // self loop
}

__device__ __forceinline__ void gat_atomic_max_f32(float* addr, float val) {
  if (val >= 0.0f) atomicMax((int*)addr, __float_as_int(val));
  else             atomicMin((unsigned int*)addr, __float_as_uint(val));
}

// e = leaky_relu(al_s[src]+al_d[dst]); store score; segment max over dst
template <int Hh>
__global__ void gat_edge_score(const long long* __restrict__ ei,
                               const float* __restrict__ als,
                               const float* __restrict__ ald,
                               float* __restrict__ escore, float* __restrict__ mbuf) {
  long long t = (long long)blockIdx.x * blockDim.x + threadIdx.x;
  if (t >= (long long)GET * Hh) return;
  int e = (int)(t / Hh), h = (int)(t & (Hh - 1));
  int s, d; gat_edge_nodes(ei, e, s, d);
  float v = als[(size_t)s * Hh + h] + ald[(size_t)d * Hh + h];
  v = v > 0.0f ? v : NEG_SLOPE * v;
  escore[t] = v;
  gat_atomic_max_f32(&mbuf[(size_t)d * Hh + h], v);
}

// w = exp(e - m[dst]); segment sum over dst
template <int Hh>
__global__ void gat_edge_exp(const long long* __restrict__ ei,
                             const float* __restrict__ mbuf,
                             float* __restrict__ escore, float* __restrict__ denom) {
  long long t = (long long)blockIdx.x * blockDim.x + threadIdx.x;
  if (t >= (long long)GET * Hh) return;
  int e = (int)(t / Hh), h = (int)(t & (Hh - 1));
  int s, d; gat_edge_nodes(ei, e, s, d);
  (void)s;
  float w = expf(escore[t] - mbuf[(size_t)d * Hh + h]);
  escore[t] = w;
  atomicAdd(&denom[(size_t)d * Hh + h], w);
}

// out[dst,h,:] += XL[src,h,:] * (w / (denom[dst,h]+1e-16)), float4 vectorized
template <int Hh, int Cc>
__global__ void gat_edge_aggregate(const long long* __restrict__ ei,
                                   const float* __restrict__ XL,
                                   const float* __restrict__ escore,
                                   const float* __restrict__ denom,
                                   float* __restrict__ outp) {
  constexpr int C4 = Cc / 4;
  long long t = (long long)blockIdx.x * blockDim.x + threadIdx.x;
  long long tot = (long long)GET * Hh * C4;
  if (t >= tot) return;
  int cq = (int)(t & (C4 - 1));
  long long r = t / C4;
  int h = (int)(r & (Hh - 1));
  int e = (int)(r / Hh);
  int s, d; gat_edge_nodes(ei, e, s, d);
  float alpha = escore[(size_t)e * Hh + h] / (denom[(size_t)d * Hh + h] + 1e-16f);
  const float4 v = *(const float4*)(XL + ((size_t)s * Hh + h) * Cc + cq * 4);
  float* op = outp + ((size_t)d * Hh + h) * Cc + cq * 4;
  atomicAdd(op + 0, v.x * alpha);
  atomicAdd(op + 1, v.y * alpha);
  atomicAdd(op + 2, v.z * alpha);
  atomicAdd(op + 3, v.w * alpha);
}

// out = (elu ? ELU : id)(out + bias)
template <int HC, int DO_ELU>
__global__ void gat_bias_act(float* __restrict__ io, const float* __restrict__ bias) {
  long long t = (long long)blockIdx.x * blockDim.x + threadIdx.x;
  if (t >= (long long)GN * HC) return;
  int j = (int)(t & (HC - 1));
  float v = io[t] + bias[j];
  if (DO_ELU) v = v > 0.0f ? v : expm1f(v);
  io[t] = v;
}

// ---------------------------------------------------------------------------
// Host-side orchestration
// ---------------------------------------------------------------------------
static inline unsigned cdivu(long long a, long long b) {
  return (unsigned)((a + b - 1) / b);
}

struct GatWs {
  float*  XL;     // [GN, 1024] max
  float*  OUTB;   // [GN, 1024] max
  bf16_t* Xb;     // [GN, 1024] max
  bf16_t* Wt;     // [1024, 1024] max (transposed weights)
  float*  als;    // [GN, 8]
  float*  ald;    // [GN, 8]
  float*  mbuf;   // [GN, 8]
  float*  den;    // [GN, 8]
  float*  wbuf;   // [GET, 8]
};

template <int Hh, int Cc, int DO_ELU>
static void gat_run_layer(const float* x_in, int K,
                          const float* W, const float* a_src, const float* a_dst,
                          const float* bias, float* out_f32,
                          const long long* ei, const GatWs& ws, hipStream_t stream) {
  constexpr int HC = Hh * Cc;
  const long long nX = (long long)GN * K;
  const long long nW = (long long)K * HC;

  // 1) bf16 conversion of activations; transpose+convert of weights
  gat_cvt_bf16<<<cdivu(nX / 4, 256), 256, 0, stream>>>(x_in, ws.Xb, nX / 4);
  gat_cvt_w_trans<<<cdivu(nW, 256), 256, 0, stream>>>(W, ws.Wt, K, HC);

  // 2) XL = Xb @ W  (WMMA bf16 -> f32), B consumed as transposed [HC,K]
  dim3 ggrid(cdivu(GN, BM), (unsigned)(HC / BN));
  gat_gemm_bf16<<<ggrid, 256, 0, stream>>>(ws.Xb, ws.Wt, ws.XL, GN, K, HC);

  // 3) attention logits
  gat_attn_logits<Hh, Cc><<<cdivu((long long)GN * Hh, 256), 256, 0, stream>>>(
      ws.XL, a_src, a_dst, ws.als, ws.ald);

  // 4) segment softmax over edges
  long long nNH = (long long)GN * Hh;
  gat_fill<<<cdivu(nNH, 256), 256, 0, stream>>>(ws.mbuf, -INFINITY, nNH);
  gat_fill<<<cdivu(nNH, 256), 256, 0, stream>>>(ws.den, 0.0f, nNH);
  long long nEH = (long long)GET * Hh;
  gat_edge_score<Hh><<<cdivu(nEH, 256), 256, 0, stream>>>(ei, ws.als, ws.ald,
                                                          ws.wbuf, ws.mbuf);
  gat_edge_exp<Hh><<<cdivu(nEH, 256), 256, 0, stream>>>(ei, ws.mbuf, ws.wbuf, ws.den);

  // 5) aggregate messages
  long long nOut = (long long)GN * HC;
  gat_fill<<<cdivu(nOut, 256), 256, 0, stream>>>(out_f32, 0.0f, nOut);
  long long nAgg = (long long)GET * Hh * (Cc / 4);
  gat_edge_aggregate<Hh, Cc><<<cdivu(nAgg, 256), 256, 0, stream>>>(
      ei, ws.XL, ws.wbuf, ws.den, out_f32);

  // 6) bias (+ ELU)
  gat_bias_act<HC, DO_ELU><<<cdivu(nOut, 256), 256, 0, stream>>>(out_f32, bias);
}

extern "C" void kernel_launch(void* const* d_in, const int* in_sizes, int n_in,
                              void* d_out, int out_size, void* d_ws, size_t ws_size,
                              hipStream_t stream) {
  (void)in_sizes; (void)n_in; (void)out_size; (void)ws_size;

  const float*     x  = (const float*)d_in[0];
  const long long* ei = (const long long*)d_in[1];
  const float *W1 = (const float*)d_in[2],  *as1 = (const float*)d_in[3],
              *ad1 = (const float*)d_in[4], *b1  = (const float*)d_in[5];
  const float *W2 = (const float*)d_in[6],  *as2 = (const float*)d_in[7],
              *ad2 = (const float*)d_in[8], *b2  = (const float*)d_in[9];
  const float *W3 = (const float*)d_in[10], *as3 = (const float*)d_in[11],
              *ad3 = (const float*)d_in[12], *b3 = (const float*)d_in[13];

  // workspace carve-up
  GatWs ws;
  float* f = (float*)d_ws;
  ws.XL   = f;                 f += (size_t)GN * 1024;
  ws.OUTB = f;                 f += (size_t)GN * 1024;
  bf16_t* bptr = (bf16_t*)f;
  ws.Xb = bptr;                bptr += (size_t)GN * 1024;
  ws.Wt = bptr;                bptr += (size_t)1024 * 1024;
  f = (float*)bptr;
  ws.als  = f; f += (size_t)GN * 8;
  ws.ald  = f; f += (size_t)GN * 8;
  ws.mbuf = f; f += (size_t)GN * 8;
  ws.den  = f; f += (size_t)GN * 8;
  ws.wbuf = f; f += (size_t)GET * 8;

  // Layer 1: in=256, H=8, C=64, concat -> 512, ELU
  gat_run_layer<8, 64, 1>(x,       256, W1, as1, ad1, b1, ws.OUTB, ei, ws, stream);
  // Layer 2: in=512, H=8, C=128, concat -> 1024, ELU  (input is consumed into
  // Xb before OUTB is reused as the output buffer)
  gat_run_layer<8, 128, 1>(ws.OUTB, 512, W2, as2, ad2, b2, ws.OUTB, ei, ws, stream);
  // Layer 3: in=1024, H=1, C=128, single head (mean == identity), no ELU
  gat_run_layer<1, 128, 0>(ws.OUTB, 1024, W3, as3, ad3, b3, (float*)d_out, ei, ws, stream);
}
